// SAModule_44693429682243
// MI455X (gfx1250) — matrix operand order repaired and checked by reading, not compiled, not run on hardware
//
#include <hip/hip_runtime.h>

#define NPTS 4096
#define MCEN 1024
#define NB   4

typedef __attribute__((ext_vector_type(16))) __bf16 v16bf;
typedef __attribute__((ext_vector_type(8)))  float  v8f;

union Frag {
    v16bf v;
    uint4 q[2];
};

__device__ inline unsigned short f2bf(float f) {
    unsigned int u = __float_as_uint(f);
    unsigned int r = (u + 0x7FFFu + ((u >> 16) & 1u)) >> 16;
    return (unsigned short)r;
}

__device__ inline v8f wmma_bf16(const Frag& a, const Frag& b, v8f c) {
    return __builtin_amdgcn_wmma_f32_16x16x32_bf16(
        false, a.v, false, b.v, (short)0, c, false, false);
}

// A fragment (16x32 bf16): lane<16 holds K runs [ks+0..7],[ks+16..23];
// lane>=16 holds [ks+8..15],[ks+24..31]; row = lane&15. Two b128 loads.
__device__ inline Frag load_a(const unsigned short* base, int lr, int hh, int ks, int stride) {
    Frag f;
    const unsigned short* r = base + lr * stride + ks + hh * 8;
    f.q[0] = *(const uint4*)(r);
    f.q[1] = *(const uint4*)(r + 16);
    return f;
}

// B fragment: pre-packed so lane's 16 halves are contiguous (32 bytes).
__device__ inline Frag load_b(const unsigned short* wp, int kt, int nt, int NT, int lane) {
    Frag f;
    const unsigned short* p = wp + (size_t)(((kt * NT + nt) * 32) + lane) * 16;
    f.q[0] = *(const uint4*)(p);
    f.q[1] = *(const uint4*)(p + 8);
    return f;
}

template <int KT, int NT>
__device__ inline void gemm_tile(const unsigned short* A, int strideA,
                                 const unsigned short* Wp, const float* bias,
                                 int lr, int hh, int lane, v8f* acc) {
#pragma unroll
    for (int nt = 0; nt < NT; ++nt) {
        float bv = bias[nt * 16 + lr];
        v8f c;
#pragma unroll
        for (int i = 0; i < 8; ++i) c[i] = bv;
        acc[nt] = c;
    }
#pragma unroll
    for (int kt = 0; kt < KT; ++kt) {
        Frag af = load_a(A, lr, hh, kt * 32, strideA);
#pragma unroll
        for (int nt = 0; nt < NT; ++nt) {
            Frag bf_ = load_b(Wp, kt, nt, NT, lane);
            acc[nt] = wmma_bf16(af, bf_, acc[nt]);
        }
    }
}

__device__ inline void relu_store(unsigned short* dst, int stride, const v8f* acc,
                                  int NT, int lr, int hh) {
    for (int nt = 0; nt < NT; ++nt) {
        int col = nt * 16 + lr;
#pragma unroll
        for (int i = 0; i < 8; ++i) {
            v8f a = acc[nt];
            dst[(hh * 8 + i) * stride + col] = f2bf(fmaxf(a[i], 0.f));
        }
    }
}

// ---------------- FPS: one block per graph ----------------
__global__ void fps_kernel(const float* __restrict__ pos, int* __restrict__ sel,
                           float* __restrict__ pdst, float* __restrict__ pos_out,
                           int* __restrict__ batch_out) {
    int b = blockIdx.x;
    int tid = threadIdx.x;  // 256 threads
    __shared__ float dist[NPTS];
    __shared__ int ssel[MCEN];
    __shared__ float bwv[8];
    __shared__ int bwi[8];
    __shared__ int s_nxt;
    __shared__ float s_p[3];

    const float* P = pos + (size_t)b * NPTS * 3;
    float x0 = P[0], y0 = P[1], z0 = P[2];
    for (int j = tid; j < NPTS; j += 256) {
        float dx = P[j * 3] - x0, dy = P[j * 3 + 1] - y0, dz = P[j * 3 + 2] - z0;
        dist[j] = dx * dx + dy * dy + dz * dz;
    }
    if (tid == 0) ssel[0] = 0;
    __syncthreads();

    for (int i = 1; i < MCEN; ++i) {
        float bd = -1.f;
        int bi = 0;
        for (int j = tid; j < NPTS; j += 256) {
            float d = dist[j];
            if (d > bd) { bd = d; bi = j; }
        }
        for (int off = 16; off > 0; off >>= 1) {
            float od = __shfl_down(bd, off, 32);
            int   oi = __shfl_down(bi, off, 32);
            if (od > bd || (od == bd && oi < bi)) { bd = od; bi = oi; }
        }
        int lane = tid & 31, wv = tid >> 5;
        if (lane == 0) { bwv[wv] = bd; bwi[wv] = bi; }
        __syncthreads();
        if (tid == 0) {
            float d = bwv[0]; int ix = bwi[0];
            for (int k = 1; k < 8; ++k)
                if (bwv[k] > d || (bwv[k] == d && bwi[k] < ix)) { d = bwv[k]; ix = bwi[k]; }
            s_nxt = ix;
            ssel[i] = ix;
            s_p[0] = P[ix * 3]; s_p[1] = P[ix * 3 + 1]; s_p[2] = P[ix * 3 + 2];
        }
        __syncthreads();
        float nx = s_p[0], ny = s_p[1], nz = s_p[2];
        (void)s_nxt;
        for (int j = tid; j < NPTS; j += 256) {
            float dx = P[j * 3] - nx, dy = P[j * 3 + 1] - ny, dz = P[j * 3 + 2] - nz;
            float d2 = dx * dx + dy * dy + dz * dz;
            dist[j] = fminf(dist[j], d2);
        }
        __syncthreads();
    }

    for (int t = tid; t < MCEN; t += 256) {
        int ix = ssel[t];
        int gc = b * MCEN + t;
        sel[gc] = ix;
        float px = P[ix * 3], py = P[ix * 3 + 1], pz = P[ix * 3 + 2];
        pdst[gc * 3] = px; pdst[gc * 3 + 1] = py; pdst[gc * 3 + 2] = pz;
        pos_out[gc * 3] = px; pos_out[gc * 3 + 1] = py; pos_out[gc * 3 + 2] = pz;
        batch_out[gc] = b;
    }
}

// ---------------- Ball query: one thread per centroid ----------------
__global__ void ballq_kernel(const float* __restrict__ pos, const float* __restrict__ pdst,
                             int* __restrict__ nbr, int K, float r2) {
    int tid = threadIdx.x;           // 64 threads
    int gc = blockIdx.x * 64 + tid;  // global centroid
    int b = gc >> 10;
    __shared__ float sd[64 * 64];
    __shared__ int   si[64 * 64];
    float* md = sd + tid * 64;
    int*   mi = si + tid * 64;

    float cx = pdst[gc * 3], cy = pdst[gc * 3 + 1], cz = pdst[gc * 3 + 2];
    const float* P = pos + (size_t)b * NPTS * 3;
    int cnt = 0;
    float worst = -1.f;
    int wslot = 0;
    for (int j = 0; j < NPTS; ++j) {
        float dx = P[j * 3] - cx, dy = P[j * 3 + 1] - cy, dz = P[j * 3 + 2] - cz;
        float d2 = dx * dx + dy * dy + dz * dz;
        if (d2 > r2) continue;
        if (cnt < K) {
            md[cnt] = d2; mi[cnt] = j;
            if (d2 > worst) { worst = d2; wslot = cnt; }
            cnt++;
        } else if (d2 < worst) {
            md[wslot] = d2; mi[wslot] = j;
            worst = -1.f;
            for (int s = 0; s < K; ++s)
                if (md[s] > worst) { worst = md[s]; wslot = s; }
        }
    }
    for (int s = 0; s < K; ++s)
        nbr[(size_t)gc * K + s] = (s < cnt) ? (b * NPTS + mi[s]) : -1;
}

// ---------------- Weight packing into WMMA B-fragment order ----------------
__global__ void pack_w(const float* __restrict__ W, unsigned short* __restrict__ out,
                       int Kreal, int NT, int N, int total) {
    int t = blockIdx.x * 256 + threadIdx.x;
    if (t >= total) return;
    int j = t & 15;
    int lane = (t >> 4) & 31;
    int g = t >> 9;          // kt*NT + nt
    int nt = g % NT;
    int kt = g / NT;
    int k = kt * 32 + (lane >> 4) * 8 + (j & 7) + (j >> 3) * 16;
    int col = nt * 16 + (lane & 15);
    float v = (k < Kreal) ? W[(size_t)k * N + col] : 0.f;
    out[t] = f2bf(v);
}

// ---------------- Fused gather + 3-layer MLP + masked max aggregation ----------
__global__ void sa_mlp_kernel(const float* __restrict__ x, const float* __restrict__ pos,
                              const float* __restrict__ pdst, const int* __restrict__ nbr,
                              const unsigned short* __restrict__ w1p,
                              const unsigned short* __restrict__ w2p,
                              const unsigned short* __restrict__ w3p,
                              const float* __restrict__ b1, const float* __restrict__ b2,
                              const float* __restrict__ b3, float* __restrict__ xout,
                              int kshift, int colOff) {
    __shared__ __align__(16) unsigned short stage[8][16 * 96];   // 24 KB
    __shared__ __align__(16) unsigned short stage2[8][16 * 64];  // 16 KB
    __shared__ float flags[128];
    __shared__ int agg[512];  // up to 4 centroids x 128 cols

    int tid = threadIdx.x;
    int lane = tid & 31, w = tid >> 5;
    int lr = lane & 15, hh = lane >> 4;
    int K = 1 << kshift;
    int numCent = 128 >> kshift;

    for (int t = tid; t < 512; t += 256) agg[t] = 0;
    __syncthreads();

    int rowBase = blockIdx.x * 128 + w * 16;

    // ---- stage h0 = [x_j | p_j - p_c | 0-pad] as bf16, 16 rows x 96 cols ----
    {
        int row = lr;
        int gr = rowBase + row;
        int cent = gr >> kshift;
        int nb = nbr[gr];
        unsigned short* st = stage[w] + row * 96;
        if (hh == 0) flags[w * 16 + row] = (nb >= 0) ? 1.f : 0.f;
        int cbase = hh * 48;
        if (nb < 0) {
            for (int c = 0; c < 48; ++c) st[cbase + c] = 0;
        } else {
            const float* xr = x + (size_t)nb * 64;
            for (int c = 0; c < 48; ++c) {
                int col = cbase + c;
                float v;
                if (col < 64)      v = xr[col];
                else if (col < 67) v = pos[(size_t)nb * 3 + (col - 64)] -
                                       pdst[(size_t)cent * 3 + (col - 64)];
                else               v = 0.f;
                st[col] = f2bf(v);
            }
        }
    }

    // ---- layer 1: 96 -> 64 ----
    v8f acc1[4];
    gemm_tile<3, 4>(stage[w], 96, w1p, b1, lr, hh, lane, acc1);
    relu_store(stage2[w], 64, acc1, 4, lr, hh);

    // ---- layer 2: 64 -> 64 (write back into stage, h0 is dead) ----
    v8f acc2[4];
    gemm_tile<2, 4>(stage2[w], 64, w2p, b2, lr, hh, lane, acc2);
    relu_store(stage[w], 96, acc2, 4, lr, hh);

    // ---- layer 3: 64 -> 128 ----
    v8f acc3[8];
    gemm_tile<2, 8>(stage[w], 96, w3p, b3, lr, hh, lane, acc3);

    // ---- masked column-max over the wave's 16 rows, then LDS atomic max ----
    int centLocal = (w * 16) >> kshift;
    for (int nt = 0; nt < 8; ++nt) {
        float m = 0.f;
#pragma unroll
        for (int i = 0; i < 8; ++i) {
            float fl = flags[w * 16 + hh * 8 + i];
            v8f a = acc3[nt];
            float v = (fl > 0.f) ? fmaxf(a[i], 0.f) : 0.f;
            m = fmaxf(m, v);
        }
        m = fmaxf(m, __shfl_xor(m, 16, 32));
        if (hh == 0)
            atomicMax(&agg[centLocal * 128 + nt * 16 + lr], __float_as_int(m));
    }
    __syncthreads();

    int centBase = (blockIdx.x * 128) >> kshift;
    (void)K;
    for (int t = tid; t < numCent * 128; t += 256) {
        int cl = t >> 7, col = t & 127;
        xout[(size_t)(centBase + cl) * 256 + colOff + col] =
            __int_as_float(agg[cl * 128 + col]);
    }
}

extern "C" void kernel_launch(void* const* d_in, const int* in_sizes, int n_in,
                              void* d_out, int out_size, void* d_ws, size_t ws_size,
                              hipStream_t stream) {
    (void)in_sizes; (void)n_in; (void)out_size; (void)ws_size;
    const float* x    = (const float*)d_in[0];
    const float* pos  = (const float*)d_in[1];
    const float* w11  = (const float*)d_in[3];
    const float* b11  = (const float*)d_in[4];
    const float* w12  = (const float*)d_in[5];
    const float* b12  = (const float*)d_in[6];
    const float* w13  = (const float*)d_in[7];
    const float* b13  = (const float*)d_in[8];
    const float* w21  = (const float*)d_in[9];
    const float* b21  = (const float*)d_in[10];
    const float* w22  = (const float*)d_in[11];
    const float* b22  = (const float*)d_in[12];
    const float* w23  = (const float*)d_in[13];
    const float* b23  = (const float*)d_in[14];

    float* xout      = (float*)d_out;                 // [4096, 256]
    float* pos_out   = xout + 4096 * 256;             // [4096, 3]
    int*   batch_out = (int*)(pos_out + 4096 * 3);    // [4096]

    // workspace carve-up (16B-aligned sections)
    char* ws = (char*)d_ws;
    int*   sel   = (int*)ws;                 ws += 4096 * sizeof(int);
    float* pdst  = (float*)ws;               ws += 4096 * 3 * sizeof(float);
    int*   nbr1  = (int*)ws;                 ws += 4096 * 32 * sizeof(int);
    int*   nbr2  = (int*)ws;                 ws += 4096 * 64 * sizeof(int);
    unsigned short* w1p_1 = (unsigned short*)ws; ws += 6144 * 2;
    unsigned short* w2p_1 = (unsigned short*)ws; ws += 4096 * 2;
    unsigned short* w3p_1 = (unsigned short*)ws; ws += 8192 * 2;
    unsigned short* w1p_2 = (unsigned short*)ws; ws += 6144 * 2;
    unsigned short* w2p_2 = (unsigned short*)ws; ws += 4096 * 2;
    unsigned short* w3p_2 = (unsigned short*)ws; ws += 8192 * 2;

    // 1) FPS (also writes pos_out / batch_out / pdst)
    fps_kernel<<<NB, 256, 0, stream>>>(pos, sel, pdst, pos_out, batch_out);

    // 2) pack weights into WMMA fragment order
    pack_w<<<(6144 + 255) / 256, 256, 0, stream>>>(w11, w1p_1, 67, 4, 64, 6144);
    pack_w<<<(4096 + 255) / 256, 256, 0, stream>>>(w12, w2p_1, 64, 4, 64, 4096);
    pack_w<<<(8192 + 255) / 256, 256, 0, stream>>>(w13, w3p_1, 64, 8, 128, 8192);
    pack_w<<<(6144 + 255) / 256, 256, 0, stream>>>(w21, w1p_2, 67, 4, 64, 6144);
    pack_w<<<(4096 + 255) / 256, 256, 0, stream>>>(w22, w2p_2, 64, 4, 64, 4096);
    pack_w<<<(8192 + 255) / 256, 256, 0, stream>>>(w23, w3p_2, 64, 8, 128, 8192);

    // 3) ball queries (r=0.2 K=32, r=0.4 K=64)
    ballq_kernel<<<4096 / 64, 64, 0, stream>>>(pos, pdst, nbr1, 32, 0.2f * 0.2f);
    ballq_kernel<<<4096 / 64, 64, 0, stream>>>(pos, pdst, nbr2, 64, 0.4f * 0.4f);

    // 4) fused gather + MLP + max-pool (WMMA bf16)
    //    branch 1: rows = 4096*32 -> 1024 blocks of 128 rows
    sa_mlp_kernel<<<1024, 256, 0, stream>>>(x, pos, pdst, nbr1, w1p_1, w2p_1, w3p_1,
                                            b11, b12, b13, xout, 5, 0);
    //    branch 2: rows = 4096*64 -> 2048 blocks
    sa_mlp_kernel<<<2048, 256, 0, stream>>>(x, pos, pdst, nbr2, w1p_2, w2p_2, w3p_2,
                                            b21, b22, b23, xout, 6, 128);
}
